// MemoryUnit_3487513444353
// MI455X (gfx1250) — compile-verified, matrix operand order checked by reading, not run on hardware
//
#include <hip/hip_runtime.h>

// ---------------------------------------------------------------------------
// MemoryUnit fused pipeline for MI455X (gfx1250, wave32, WMMA).
// Precision choice: bf16 WMMA (16x16x32, f32 accum) for all GEMMs -> the
// kernel stays HBM-bound (~1.35 GB traffic @ 23.3 TB/s ~= 58us) instead of
// fp32-WMMA compute-bound. W (320KB bf16) stays hot in 192MB L2.
// ---------------------------------------------------------------------------

#define NUM_CLS  10
#define PTT_NUM  8
#define PART_NUM 4
#define FEA_DIM  512
#define MEM_DIM  320
#define NROWS    131072
#define SHRINKL  0.0025f
#define EPSF     1e-12f

#define TM 128          // rows per workgroup
#define NTHREADS 256    // 8 waves of 32

typedef __attribute__((ext_vector_type(16))) __bf16    v16bf;
typedef __attribute__((ext_vector_type(8)))  float     v8f;
typedef __attribute__((ext_vector_type(4)))  unsigned  u32x4;
typedef __attribute__((ext_vector_type(4)))  float     f32x4;

union FragU {           // 32 bytes = 8 VGPRs = one 16x32 bf16 A/B fragment
    u32x4 u[2];
    v16bf v;
};

__device__ __forceinline__ unsigned short f2bf(float f) {
    union { float f; unsigned u; } v; v.f = f;
    unsigned u = v.u;
    u = u + 0x7FFFu + ((u >> 16) & 1u);   // round-to-nearest-even
    return (unsigned short)(u >> 16);
}
__device__ __forceinline__ unsigned pack_bf2(float a, float b) {
    return (unsigned)f2bf(a) | ((unsigned)f2bf(b) << 16);
}

// ---------------------------------------------------------------------------
// Kernel A: sem_att (gating network on W) + bf16 conversions of W.
//   Wbf  [320,512] row-major  : B operand for logits GEMM (rows of W, contig)
//   WbfT [512,320] row-major  : B operand for readout GEMM (cols of W, contig)
//   sem_bf  [16,512] (rows>=10 zero) : B operand for sem-logits GEMM
//   sem_bfT [512,32] (cols>=10 zero) : B operand for sem-readout GEMM (K pad 32)
// ---------------------------------------------------------------------------
__global__ void __launch_bounds__(512) setup_kernel(
    const float* __restrict__ W,  const float* __restrict__ wp,
    const float* __restrict__ bp, const float* __restrict__ wi,
    const float* __restrict__ bi, float* __restrict__ out_sem_att,
    unsigned short* __restrict__ Wbf,    unsigned short* __restrict__ WbfT,
    unsigned short* __restrict__ sem_bf, unsigned short* __restrict__ sem_bfT)
{
    const int c = threadIdx.x;            // 0..511 : one feature channel each
    float wp_[16], bp_[4], wi_[64], bi_[8];
#pragma unroll
    for (int i = 0; i < 16; ++i) wp_[i] = wp[i];
#pragma unroll
    for (int i = 0; i < 4;  ++i) bp_[i] = bp[i];
#pragma unroll
    for (int i = 0; i < 64; ++i) wi_[i] = wi[i];
#pragma unroll
    for (int i = 0; i < 8;  ++i) bi_[i] = bi[i];

    float sem[NUM_CLS];
#pragma unroll
    for (int cls = 0; cls < NUM_CLS; ++cls) {
        float pia[PTT_NUM];
#pragma unroll
        for (int t8 = 0; t8 < PTT_NUM; ++t8) {
            const int l = cls * PTT_NUM + t8;
            // rp[l][p][c] = W_flat[l*2048 + c*4 + p]  -> one float4
            f32x4 r = ((const f32x4*)W)[l * 512 + c];
            float rp[4] = { r.x, r.y, r.z, r.w };
            float acc = 0.f;
#pragma unroll
            for (int q = 0; q < 4; ++q) {
                float z = wp_[q*4+0]*rp[0] + wp_[q*4+1]*rp[1] +
                          wp_[q*4+2]*rp[2] + wp_[q*4+3]*rp[3] + bp_[q];
                float g = 1.f / (1.f + __expf(-z));
                acc += g * rp[q];                 // (gp * rp) then mean over q
            }
            pia[t8] = 0.25f * acc;
        }
        float a2 = 0.f;
#pragma unroll
        for (int q = 0; q < 8; ++q) {
            float z = bi_[q];
#pragma unroll
            for (int tt = 0; tt < 8; ++tt) z += wi_[q*8+tt] * pia[tt];
            float g = 1.f / (1.f + __expf(-z));
            a2 += g * pia[q];
        }
        sem[cls] = 0.125f * a2;
    }

#pragma unroll
    for (int cls = 0; cls < NUM_CLS; ++cls) out_sem_att[cls*512 + c] = sem[cls];
#pragma unroll
    for (int l = 0; l < 16; ++l)
        sem_bf[l*512 + c] = (l < NUM_CLS) ? f2bf(sem[l]) : (unsigned short)0;
#pragma unroll
    for (int j = 0; j < 32; ++j)
        sem_bfT[c*32 + j] = (j < NUM_CLS) ? f2bf(sem[j]) : (unsigned short)0;

    for (int n = 0; n < MEM_DIM; ++n) {
        float w = W[n*512 + c];
        unsigned short b = f2bf(w);
        Wbf [n*512 + c] = b;
        WbfT[c*320 + n] = b;
    }
}

// ---------------------------------------------------------------------------
// Kernel B: fused attend pipeline, one 128-row tile per workgroup.
// LDS layout (dynamic, 262144 bytes):
//   [0, 131072)        input tile bf16 [128,512]  (phase 1-2 only)
//   [0, 163840)        logits f32 [128,320]       (after barrier, overlaps)
//   [163840, 245760)   att bf16 [128,320]
//   [245760, 253952)   sem logits f32 [128,16]
//   [253952, 262144)   att_sem bf16 [128,32] (K padded to 32)
// ---------------------------------------------------------------------------
__global__ void __launch_bounds__(NTHREADS) memunit_kernel(
    const float* __restrict__ input,
    const unsigned short* __restrict__ Wbf,
    const unsigned short* __restrict__ WbfT,
    const unsigned short* __restrict__ sem_bf,
    const unsigned short* __restrict__ sem_bfT,
    float* __restrict__ out_sem,    // [N,512]
    float* __restrict__ out_att,    // [N,10]
    float* __restrict__ out_part)   // [N,512]
{
    extern __shared__ char smem[];
    float*          logits  = (float*)smem;
    unsigned*       inp_bf  = (unsigned*)smem;
    unsigned short* att_bf  = (unsigned short*)(smem + 163840);
    float*          slog    = (float*)(smem + 245760);
    unsigned short* asem_bf = (unsigned short*)(smem + 253952);

    const int  t    = threadIdx.x;
    const int  wave = t >> 5;
    const int  lane = t & 31;
    const int  l16  = lane & 15;
    const int  half = lane >> 4;
    const long rowBase = (long)blockIdx.x * TM;

    // ---- phase 1: coalesced global load of input tile -> bf16 LDS ----
    {
        const f32x4* src = (const f32x4*)(input + rowBase * FEA_DIM);
        for (int i = t; i < TM * FEA_DIM / 4; i += NTHREADS) {
            f32x4 v = src[i];
            inp_bf[i*2 + 0] = pack_bf2(v.x, v.y);
            inp_bf[i*2 + 1] = pack_bf2(v.z, v.w);
        }
    }
    __syncthreads();

    // ---- phase 2: build 16 input A-fragments (CDNA5 16-bit A layout) ----
    // lane<16 holds row l16, K = {kb..kb+7, kb+16..kb+23};
    // lane>=16 holds row l16, K = {kb+8..kb+15, kb+24..kb+31}.
    FragU af[16];
    {
        const u32x4* ib = (const u32x4*)smem;       // bf16 row stride = 64 u32x4
        const int arow = wave*16 + l16;
#pragma unroll
        for (int ks = 0; ks < 16; ++ks) {
            const int idx = arow*64 + ks*4 + half;
            af[ks].u[0] = ib[idx];
            af[ks].u[1] = ib[idx + 2];
        }
    }
    __syncthreads();                                // logits overwrite input region

    // ---- phase 3: level-1 logits [128,320] + sem logits [128,16] ----
    {
        const u32x4* wb = (const u32x4*)Wbf;        // row stride 64 u32x4
        for (int nt = 0; nt < MEM_DIM/16; ++nt) {
            const int n = nt*16 + l16;              // B column = W row (B=W^T)
            v8f acc = {};
#pragma unroll
            for (int ks = 0; ks < 16; ++ks) {
                FragU b;
                const int idx = n*64 + ks*4 + half;
                b.u[0] = wb[idx];
                b.u[1] = wb[idx + 2];
                acc = __builtin_amdgcn_wmma_f32_16x16x32_bf16(
                        false, af[ks].v, false, b.v, (short)0, acc, false, false);
            }
#pragma unroll
            for (int v = 0; v < 8; ++v) {
                const int m = wave*16 + v + 8*half;
                logits[m*MEM_DIM + nt*16 + l16] = acc[v];
            }
        }
        {   // sem logits: B = sem_att^T (rows >= 10 are zero)
            const u32x4* sb = (const u32x4*)sem_bf;
            v8f acc = {};
#pragma unroll
            for (int ks = 0; ks < 16; ++ks) {
                FragU b;
                const int idx = l16*64 + ks*4 + half;
                b.u[0] = sb[idx];
                b.u[1] = sb[idx + 2];
                acc = __builtin_amdgcn_wmma_f32_16x16x32_bf16(
                        false, af[ks].v, false, b.v, (short)0, acc, false, false);
            }
#pragma unroll
            for (int v = 0; v < 8; ++v) {
                const int m = wave*16 + v + 8*half;
                slog[m*16 + l16] = acc[v];
            }
        }
    }
    __syncthreads();

    // ---- phase 4: softmax -> hard-shrink-relu -> L1 normalize (per row) ----
    if (t < TM) {
        float* lr = logits + t * MEM_DIM;
        float mx = -3.4e38f;
        for (int j = 0; j < MEM_DIM; ++j) mx = fmaxf(mx, lr[j]);
        float s = 0.f;
        for (int j = 0; j < MEM_DIM; ++j) { float e = __expf(lr[j]-mx); lr[j] = e; s += e; }
        const float inv = 1.f / s;
        float s2 = 0.f;
        for (int j = 0; j < MEM_DIM; ++j) {
            float a = lr[j] * inv;
            float h = fmaxf(a - SHRINKL, 0.f) * a / (fabsf(a - SHRINKL) + EPSF);
            lr[j] = h; s2 += fabsf(h);
        }
        const float n2 = 1.f / fmaxf(s2, EPSF);
        for (int j = 0; j < MEM_DIM; ++j) att_bf[t*MEM_DIM + j] = f2bf(lr[j] * n2);

        // level-3 attention over 10 classes
        float* sr = slog + t*16;
        float smx = -3.4e38f;
#pragma unroll
        for (int j = 0; j < NUM_CLS; ++j) smx = fmaxf(smx, sr[j]);
        float ev[NUM_CLS]; float ss = 0.f;
#pragma unroll
        for (int j = 0; j < NUM_CLS; ++j) { ev[j] = __expf(sr[j]-smx); ss += ev[j]; }
        const float sinv = 1.f / ss;
        float ss2 = 0.f;
#pragma unroll
        for (int j = 0; j < NUM_CLS; ++j) {
            float a = ev[j] * sinv;
            float h = fmaxf(a - SHRINKL, 0.f) * a / (fabsf(a - SHRINKL) + EPSF);
            ev[j] = h; ss2 += fabsf(h);
        }
        const float sn2 = 1.f / fmaxf(ss2, EPSF);
        const long grow = rowBase + t;
#pragma unroll
        for (int j = 0; j < NUM_CLS; ++j) {
            const float a = ev[j] * sn2;
            out_att[grow*NUM_CLS + j] = a;      // f32 att output
            asem_bf[t*32 + j] = f2bf(a);
        }
#pragma unroll
        for (int j = NUM_CLS; j < 32; ++j) asem_bf[t*32 + j] = 0;
    }
    __syncthreads();

    // ---- phase 5: readouts: output_part = att@W, output_sem = att_sem@sem_att
    FragU aatt[10];
    {
        const u32x4* ab = (const u32x4*)att_bf;     // row stride 40 u32x4
        const int arow = wave*16 + l16;
#pragma unroll
        for (int ks = 0; ks < 10; ++ks) {
            const int idx = arow*40 + ks*4 + half;
            aatt[ks].u[0] = ab[idx];
            aatt[ks].u[1] = ab[idx + 2];
        }
    }
    FragU asem;
    {
        const u32x4* ab = (const u32x4*)asem_bf;    // row stride 4 u32x4
        const int arow = wave*16 + l16;
        asem.u[0] = ab[arow*4 + half];
        asem.u[1] = ab[arow*4 + half + 2];
    }
    {
        const u32x4* wtb = (const u32x4*)WbfT;      // row stride 40 u32x4
        const u32x4* stb = (const u32x4*)sem_bfT;   // row stride 4 u32x4
        for (int ct = 0; ct < FEA_DIM/16; ++ct) {
            const int c = ct*16 + l16;
            v8f acc = {};
#pragma unroll
            for (int ks = 0; ks < 10; ++ks) {       // K = 320
                FragU b;
                const int idx = c*40 + ks*4 + half;
                b.u[0] = wtb[idx];
                b.u[1] = wtb[idx + 2];
                acc = __builtin_amdgcn_wmma_f32_16x16x32_bf16(
                        false, aatt[ks].v, false, b.v, (short)0, acc, false, false);
            }
            v8f acc2 = {};
            {
                FragU b;
                b.u[0] = stb[c*4 + half];
                b.u[1] = stb[c*4 + half + 2];
                acc2 = __builtin_amdgcn_wmma_f32_16x16x32_bf16(
                        false, asem.v, false, b.v, (short)0, acc2, false, false);
            }
#pragma unroll
            for (int v = 0; v < 8; ++v) {
                const long m = rowBase + wave*16 + v + 8*half;
                out_part[m*FEA_DIM + ct*16 + l16] = acc[v];
                out_sem [m*FEA_DIM + ct*16 + l16] = acc2[v];
            }
        }
    }
}

// ---------------------------------------------------------------------------
extern "C" void kernel_launch(void* const* d_in, const int* in_sizes, int n_in,
                              void* d_out, int out_size, void* d_ws, size_t ws_size,
                              hipStream_t stream) {
    const float* input = (const float*)d_in[0];
    const float* W     = (const float*)d_in[1];
    const float* wp    = (const float*)d_in[2];
    const float* bp    = (const float*)d_in[3];
    const float* wi    = (const float*)d_in[4];
    const float* bi    = (const float*)d_in[5];

    float* out = (float*)d_out;
    const long N = NROWS;
    float* out_sem    = out;                                    // [N,512]
    float* out_att    = out + N*(long)FEA_DIM;                  // [N,10]
    float* out_sematt = out_att + N*(long)NUM_CLS;              // [10,512]
    float* out_part   = out_sematt + (long)NUM_CLS*FEA_DIM;     // [N,512]

    char* ws = (char*)d_ws;
    unsigned short* Wbf     = (unsigned short*)(ws);            // 320*512*2 = 327680
    unsigned short* WbfT    = (unsigned short*)(ws + 327680);   // 512*320*2 = 327680
    unsigned short* sem_bf  = (unsigned short*)(ws + 655360);   // 16*512*2  = 16384
    unsigned short* sem_bfT = (unsigned short*)(ws + 671744);   // 512*32*2  = 32768

    setup_kernel<<<1, 512, 0, stream>>>(W, wp, bp, wi, bi,
                                        out_sematt, Wbf, WbfT, sem_bf, sem_bfT);

    const size_t smemBytes = 262144;   // 256KB of the 320KB/WGP LDS
    memunit_kernel<<<dim3(NROWS / TM), dim3(NTHREADS), smemBytes, stream>>>(
        input, Wbf, WbfT, sem_bf, sem_bfT, out_sem, out_att, out_part);
}
